// Encoder_24446953849217
// MI455X (gfx1250) — compile-verified, matrix operand order checked by reading, not run on hardware
//
#include <hip/hip_runtime.h>
#include <hip/hip_bf16.h>

// ---------------------------------------------------------------------------
// Bidirectional GRU encoder for MI455X (gfx1250, wave32, WMMA f32=f16xf16)
//
//   tokens[128,256] -> emb[32000,512] -> xp = x@Wx + b  (big WMMA GEMM)
//   256 sequential steps/dir:  rec = h@Wh (+b_rec), gates, h' (fused WMMA)
//   hidden = concat(h_f,h_b) @ fc_W + fc_b               (WMMA GEMM)
// ---------------------------------------------------------------------------

typedef __attribute__((ext_vector_type(16))) _Float16 v16h;
typedef __attribute__((ext_vector_type(8)))  _Float16 v8h;
typedef __attribute__((ext_vector_type(8)))  float    v8f;

#define B_   128
#define T_   256
#define EMB_ 512
#define U_   1024
#define G3_  3072   // 3*U
#define M_   (B_*T_)  // 32768

// ---------------- WMMA helpers -------------------------------------------

__device__ __forceinline__ v8f wmma_f16(v16h a, v16h b, v8f c) {
  return __builtin_amdgcn_wmma_f32_16x16x32_f16(
      /*neg_a=*/false, a, /*neg_b=*/false, b,
      /*c_mod=*/(short)0, c, /*reuse_a=*/false, /*reuse_b=*/false);
}

// A fragment (16x32 f16, MxK) from a row-major matrix with leading dim `ld`
// (halves). Lane l<16 holds row m0+l, K chunks [k0..k0+7] and [k0+16..k0+23];
// lane l>=16 holds row m0+(l-16), K chunks [k0+8..] and [k0+24..].
__device__ __forceinline__ v16h frag_a_rowmajor(const _Float16* base, int ld,
                                                int m0, int k0) {
  int lane = threadIdx.x & 31;
  int mrow = m0 + (lane & 15);
  int kb   = k0 + ((lane & 16) ? 8 : 0);
  const _Float16* p = base + (size_t)mrow * ld + kb;
  v8h lo = *(const v8h*)(p);
  v8h hi = *(const v8h*)(p + 16);
  v16h a;
#pragma unroll
  for (int i = 0; i < 8; ++i) { a[i] = lo[i]; a[i + 8] = hi[i]; }
  return a;
}

// B fragment (32x16 f16, KxN) from pre-packed layout:
//   packed[(((nTile*KT)+kTile)*32 + lane)*16 + h]
__device__ __forceinline__ v16h frag_b_packed(const _Float16* packed, int KT,
                                              int kTile, int nTile) {
  int lane = threadIdx.x & 31;
  const _Float16* p = packed + ((((size_t)nTile * KT + kTile) * 32 + lane) * 16);
  v8h lo = *(const v8h*)(p);
  v8h hi = *(const v8h*)(p + 8);
  v16h b;
#pragma unroll
  for (int i = 0; i < 8; ++i) { b[i] = lo[i]; b[i + 8] = hi[i]; }
  return b;
}

// Fast transcendentals on the serialized recurrence critical path:
// v_exp_f32 computes 2^x, v_rcp_f32 is a single TRANS32 op.
#define LOG2E_ 1.4426950408889634f
__device__ __forceinline__ float fast_sigmoid(float x) {
  return __builtin_amdgcn_rcpf(1.0f + __builtin_amdgcn_exp2f(-LOG2E_ * x));
}
__device__ __forceinline__ float fast_tanh(float x) {
  // tanh(x) = 2*sigmoid(2x) - 1
  return 2.0f * __builtin_amdgcn_rcpf(
                    1.0f + __builtin_amdgcn_exp2f(-2.0f * LOG2E_ * x)) -
         1.0f;
}

// ---------------- utility kernels ----------------------------------------

__global__ void zero_u32_kernel(unsigned int* p, int n) {
  int i = blockIdx.x * 256 + threadIdx.x;
  if (i < n) p[i] = 0u;
}

__global__ void f32_to_f16_kernel(const float* src, _Float16* dst, int n) {
  int i = blockIdx.x * 256 + threadIdx.x;
  if (i < n) dst[i] = (_Float16)src[i];
}

// Pack row-major f32 [K,N] into WMMA-B fragment order (f16).
__global__ void pack_b_kernel(const float* src, _Float16* dst, int K, int N) {
  size_t idx = (size_t)blockIdx.x * 256 + threadIdx.x;
  size_t total = (size_t)K * N;
  if (idx >= total) return;
  int KT = K >> 5;
  size_t tile = idx >> 9;     // 512 halves per 32x16 tile
  int r    = (int)(idx & 511);
  int lane = r >> 4;
  int h    = r & 15;
  int kTile = (int)(tile % KT);
  int nTile = (int)(tile / KT);
  int n  = nTile * 16 + (lane & 15);
  int kk = ((h < 8) ? h : h + 8) + ((lane & 16) ? 8 : 0);
  int k  = kTile * 32 + kk;
  dst[idx] = (_Float16)src[(size_t)k * N + n];
}

// ---------------- input projection: xp = emb[tokens] @ Wx + b[0] ----------
// grid: (M/16, 24, 2dirs), block 256 (8 waves). Wave -> one 16x16 tile.
__global__ void proj_kernel(const int* __restrict__ tokens,
                            const _Float16* __restrict__ embF16,
                            const _Float16* __restrict__ wxPF,
                            const _Float16* __restrict__ wxPB,
                            const float* __restrict__ bF,
                            const float* __restrict__ bB,
                            _Float16* __restrict__ xpF,
                            _Float16* __restrict__ xpB) {
  int dir = blockIdx.z;
  const _Float16* wx  = dir ? wxPB : wxPF;
  const float*    bias = dir ? bB  : bF;     // row 0 = input bias
  _Float16*       xp  = dir ? xpB : xpF;

  int wave = threadIdx.x >> 5;
  int lane = threadIdx.x & 31;
  int m0 = blockIdx.x * 16;
  int nTile = blockIdx.y * 8 + wave;           // 0..191

  int mrow = m0 + (lane & 15);
  int token = tokens[mrow];
  const _Float16* arow = embF16 + (size_t)token * EMB_;

  v8f acc = {};
#pragma unroll 4
  for (int kt = 0; kt < EMB_ / 32; ++kt) {
    int kb = kt * 32 + ((lane & 16) ? 8 : 0);
    v8h lo = *(const v8h*)(arow + kb);
    v8h hi = *(const v8h*)(arow + kb + 16);
    v16h a;
#pragma unroll
    for (int i = 0; i < 8; ++i) { a[i] = lo[i]; a[i + 8] = hi[i]; }
    v16h b = frag_b_packed(wx, EMB_ / 32, kt, nTile);
    acc = wmma_f16(a, b, acc);
  }

  int col = nTile * 16 + (lane & 15);
  float bc = bias[col];
  int rbase = (lane & 16) ? 8 : 0;
#pragma unroll
  for (int r = 0; r < 8; ++r) {
    int m = m0 + rbase + r;
    xp[(size_t)m * G3_ + col] = (_Float16)(acc[r] + bc);
  }
}

// ---------------- fused GRU step: rec = h@Wh, gates, h' -------------------
// grid: (8 rowblocks, 8 colgroups, 2 dirs), block 256 (8 waves).
// Wave -> one 16-row x 16-unit tile with 3 gate accumulation chains.
__global__ void gru_step_kernel(int s,
                                const int* __restrict__ tokens,
                                const _Float16* __restrict__ whPF,
                                const _Float16* __restrict__ whPB,
                                const float* __restrict__ bF,
                                const float* __restrict__ bB,
                                const _Float16* __restrict__ xpF,
                                const _Float16* __restrict__ xpB,
                                float* __restrict__ hF32, float* __restrict__ hB32,
                                _Float16* __restrict__ hF16, _Float16* __restrict__ hB16,
                                float* __restrict__ out) {
  int dir = blockIdx.z;
  int t = dir ? (T_ - 1 - s) : s;
  const _Float16* wh   = dir ? whPB : whPF;
  const float*    brec = (dir ? bB : bF) + G3_;   // row 1 = recurrent bias
  const _Float16* xp   = dir ? xpB : xpF;
  float*     h32 = dir ? hB32 : hF32;
  _Float16*  h16 = dir ? hB16 : hF16;

  const size_t HS = (size_t)B_ * U_;
  int pr = s & 1, pw = pr ^ 1;                    // double buffer
  const _Float16* hin   = h16 + (size_t)pr * HS;
  const float*    hin32 = h32 + (size_t)pr * HS;
  _Float16* hout16 = h16 + (size_t)pw * HS;
  float*    hout32 = h32 + (size_t)pw * HS;

  int wave = threadIdx.x >> 5;
  int lane = threadIdx.x & 31;
  int m0 = blockIdx.x * 16;
  int uTile = blockIdx.y * 8 + wave;              // 0..63
  const int KT = U_ / 32;                         // 32

  // L2-resident Wh fragment streams for this wave's three gate columns.
  __builtin_prefetch(wh + (size_t)(uTile)       * KT * 512, 0, 1);
  __builtin_prefetch(wh + (size_t)(64  + uTile) * KT * 512, 0, 1);
  __builtin_prefetch(wh + (size_t)(128 + uTile) * KT * 512, 0, 1);

  // Stage the shared 16-row h tile (16x1024 f16 = 32KB) in LDS once; all
  // 8 waves in the block reuse the same A fragments.
  __shared__ _Float16 shA[16 * U_];
  {
    const uint4* g4 = (const uint4*)(hin + (size_t)m0 * U_);
    uint4* s4 = (uint4*)shA;
    for (int i = threadIdx.x; i < 16 * U_ / 8; i += 256) s4[i] = g4[i];
  }
  __syncthreads();

  v8f accZ = {}, accR = {}, accH = {};
#pragma unroll 4
  for (int kt = 0; kt < KT; ++kt) {
    v16h a  = frag_a_rowmajor(shA, U_, 0, kt * 32);
    v16h bz = frag_b_packed(wh, KT, kt, uTile);
    v16h br = frag_b_packed(wh, KT, kt, 64 + uTile);
    v16h bh = frag_b_packed(wh, KT, kt, 128 + uTile);
    accZ = wmma_f16(a, bz, accZ);
    accR = wmma_f16(a, br, accR);
    accH = wmma_f16(a, bh, accH);
  }

  int u = uTile * 16 + (lane & 15);
  float bz = brec[u], br = brec[U_ + u], bh = brec[2 * U_ + u];
  int rbase = (lane & 16) ? 8 : 0;
#pragma unroll
  for (int r = 0; r < 8; ++r) {
    int row = m0 + rbase + r;
    size_t mt = (size_t)row * T_ + t;
    bool msk = tokens[mt] != 0;
    const _Float16* xprow = xp + mt * G3_;
    float xz = (float)xprow[u];
    float xr = (float)xprow[U_ + u];
    float xh = (float)xprow[2 * U_ + u];
    float z  = fast_sigmoid(xz + accZ[r] + bz);
    float rg = fast_sigmoid(xr + accR[r] + br);
    float hh = fast_tanh(xh + rg * (accH[r] + bh));
    float hp = hin32[(size_t)row * U_ + u];
    float hn = z * hp + (1.0f - z) * hh;
    hn = msk ? hn : hp;
    hout32[(size_t)row * U_ + u] = hn;
    hout16[(size_t)row * U_ + u] = (_Float16)hn;
    out[mt * (2 * U_) + dir * U_ + u] = hn;
  }
}

// ---------------- final FC: hidden = concat(h_f,h_b) @ fc_W + fc_b --------
// grid: (8 rowblocks, 8 colgroups), block 256 (8 waves).
__global__ void fc_kernel(const _Float16* __restrict__ hF16,
                          const _Float16* __restrict__ hB16,
                          const _Float16* __restrict__ fcP,
                          const float* __restrict__ fcB,
                          float* __restrict__ out) {
  int wave = threadIdx.x >> 5;
  int lane = threadIdx.x & 31;
  int m0 = blockIdx.x * 16;
  int nTile = blockIdx.y * 8 + wave;              // 0..63
  const int KT = (2 * U_) / 32;                   // 64

  v8f acc = {};
#pragma unroll 4
  for (int kt = 0; kt < KT; ++kt) {
    const _Float16* src = (kt < 32) ? hF16 : hB16;
    int k0 = (kt < 32) ? kt * 32 : (kt - 32) * 32;
    v16h a = frag_a_rowmajor(src, U_, m0, k0);
    v16h b = frag_b_packed(fcP, KT, kt, nTile);
    acc = wmma_f16(a, b, acc);
  }

  int col = nTile * 16 + (lane & 15);
  float bc = fcB[col];
  int rbase = (lane & 16) ? 8 : 0;
#pragma unroll
  for (int r = 0; r < 8; ++r) {
    int row = m0 + rbase + r;
    out[(size_t)row * U_ + col] = acc[r] + bc;
  }
}

// ---------------- host side ------------------------------------------------

extern "C" void kernel_launch(void* const* d_in, const int* in_sizes, int n_in,
                              void* d_out, int out_size, void* d_ws, size_t ws_size,
                              hipStream_t stream) {
  const int*   tokens = (const int*)  d_in[0];
  const float* emb    = (const float*)d_in[1];
  const float* Wx_f   = (const float*)d_in[2];
  const float* Wh_f   = (const float*)d_in[3];
  const float* b_f    = (const float*)d_in[4];
  const float* Wx_b   = (const float*)d_in[5];
  const float* Wh_b   = (const float*)d_in[6];
  const float* b_b    = (const float*)d_in[7];
  const float* fc_W   = (const float*)d_in[8];
  const float* fc_b   = (const float*)d_in[9];
  float* out = (float*)d_out;

  char* ws = (char*)d_ws;
  size_t off = 0;
  auto alloc = [&](size_t bytes) -> void* {
    void* p = ws + off;
    off = (off + bytes + 255) & ~(size_t)255;
    return p;
  };

  _Float16* embF16 = (_Float16*)alloc((size_t)32000 * EMB_ * 2);
  _Float16* wxPF   = (_Float16*)alloc((size_t)EMB_ * G3_ * 2);
  _Float16* wxPB   = (_Float16*)alloc((size_t)EMB_ * G3_ * 2);
  _Float16* whPF   = (_Float16*)alloc((size_t)U_ * G3_ * 2);
  _Float16* whPB   = (_Float16*)alloc((size_t)U_ * G3_ * 2);
  _Float16* fcP    = (_Float16*)alloc((size_t)(2 * U_) * U_ * 2);
  _Float16* xpF    = (_Float16*)alloc((size_t)M_ * G3_ * 2);
  _Float16* xpB    = (_Float16*)alloc((size_t)M_ * G3_ * 2);
  // state buffers (contiguous, zeroed together): 2x double-buffered per dir
  float*    hF32   = (float*)   alloc(2 * (size_t)B_ * U_ * 4);
  float*    hB32   = (float*)   alloc(2 * (size_t)B_ * U_ * 4);
  _Float16* hF16   = (_Float16*)alloc(2 * (size_t)B_ * U_ * 2);
  _Float16* hB16   = (_Float16*)alloc(2 * (size_t)B_ * U_ * 2);

  // 1) zero h state (f32 x2dirs x2bufs + f16 x2dirs x2bufs, contiguous)
  {
    int words = (int)((2 * B_ * U_ * 4 * 2 + 2 * B_ * U_ * 2 * 2) / 4);
    zero_u32_kernel<<<(words + 255) / 256, 256, 0, stream>>>(
        (unsigned int*)hF32, words);
  }

  // 2) precision conversion + WMMA-B fragment packing (once per call)
  {
    int n = 32000 * EMB_;
    f32_to_f16_kernel<<<(n + 255) / 256, 256, 0, stream>>>(emb, embF16, n);
  }
  pack_b_kernel<<<(EMB_ * G3_ + 255) / 256, 256, 0, stream>>>(Wx_f, wxPF, EMB_, G3_);
  pack_b_kernel<<<(EMB_ * G3_ + 255) / 256, 256, 0, stream>>>(Wx_b, wxPB, EMB_, G3_);
  pack_b_kernel<<<(U_ * G3_ + 255) / 256, 256, 0, stream>>>(Wh_f, whPF, U_, G3_);
  pack_b_kernel<<<(U_ * G3_ + 255) / 256, 256, 0, stream>>>(Wh_b, whPB, U_, G3_);
  pack_b_kernel<<<(2 * U_ * U_ + 255) / 256, 256, 0, stream>>>(fc_W, fcP, 2 * U_, U_);

  // 3) input projections for both directions (embedding gather fused)
  proj_kernel<<<dim3(M_ / 16, (G3_ / 16) / 8, 2), 256, 0, stream>>>(
      tokens, embF16, wxPF, wxPB, b_f, b_b, xpF, xpB);

  // 4) 256 fused recurrence steps, both directions per launch
  for (int s = 0; s < T_; ++s) {
    gru_step_kernel<<<dim3(B_ / 16, (U_ / 16) / 8, 2), 256, 0, stream>>>(
        s, tokens, whPF, whPB, b_f, b_b, xpF, xpB,
        hF32, hB32, hF16, hB16, out);
  }

  // 5) final FC on last hidden states (after 256 steps state is in buffer 0)
  fc_kernel<<<dim3(B_ / 16, (U_ / 16) / 8), 256, 0, stream>>>(
      hF16, hB16, fcP, fc_b, out + (size_t)B_ * T_ * 2 * U_);
}
